// PropertyPrediction_35218731827261
// MI455X (gfx1250) — matrix operand order repaired and checked by reading, not compiled, not run on hardware
//
#include <hip/hip_runtime.h>
#include <hip/hip_bf16.h>
#include <stdint.h>

typedef __bf16 bf16_t;
typedef bf16_t v16bf __attribute__((ext_vector_type(16)));
typedef float  v8f   __attribute__((ext_vector_type(8)));

typedef unsigned int tdm_v4u __attribute__((ext_vector_type(4)));
typedef int          tdm_v8i __attribute__((ext_vector_type(8)));
typedef int          tdm_v4i __attribute__((ext_vector_type(4)));

// ---------------- workspace layout (all offsets 256B aligned) ----------------
static constexpr size_t CNT_OFF  = 0;                          // 4B barrier counter
static constexpr size_t BIAS_OFF = 256;                        // 2048 f32 (b_ih+b_hh)
static constexpr size_t WG_OFF   = BIAS_OFF + 2048 * 4;        // [2048][640] bf16 = [W_ih ; W_hh]
static constexpr size_t WOUT_OFF = WG_OFF + 2048 * 640 * 2;    // [128][512] bf16
static constexpr size_t XT_OFF   = WOUT_OFF + 128 * 512 * 2;   // [512][64][128] bf16
static constexpr size_t HB_OFF   = XT_OFF + 512 * 64 * 128 * 2;// 2 x [64][512] bf16 (ping-pong h)

static constexpr int GRID_BLKS = 32;        // 32 blocks (hidden tiles) x 4 waves (batch tiles)
static constexpr int SLAB_BYTES = 64 * 640 * 2;  // 80 KB per-block weight slab in LDS

// ---------------- WMMA fragment helpers (wave32) ----------------
union FragU { v16bf v; uint4 q[2]; };

// A: 16x32 bf16, lane holds row M=lane%16, 16 consecutive K from k0 + (lane/16)*16
__device__ __forceinline__ v16bf load_frag_a(const bf16_t* __restrict__ base, int stride,
                                             int mbase, int k0, int lrow, int lhi) {
  FragU f;
  const bf16_t* p = base + (size_t)(mbase + lrow) * stride + (k0 + lhi * 16);
  f.q[0] = *reinterpret_cast<const uint4*>(p);
  f.q[1] = *reinterpret_cast<const uint4*>(p + 8);
  return f.v;
}

// B: 32x16 bf16, lane holds column N=lane%16, 16 consecutive K from k0 + (lane/16)*16
__device__ __forceinline__ v16bf load_frag_b(const bf16_t* __restrict__ base, int stride,
                                             int nbase, int k0, int lrow, int lhi) {
  FragU f;
  const bf16_t* p = base + (size_t)(nbase + lrow) * stride + (k0 + lhi * 16);
  f.q[0] = *reinterpret_cast<const uint4*>(p);
  f.q[1] = *reinterpret_cast<const uint4*>(p + 8);
  return f.v;
}

// B fragment from the LDS weight slab: slab row = gate*16 + lrow, row-major [64][640] bf16
__device__ __forceinline__ v16bf load_frag_b_lds(const bf16_t* slab, int gate, int k0,
                                                 int lrow, int lhi) {
  FragU f;
  const bf16_t* p = slab + ((gate * 16 + lrow) * 640 + k0 + lhi * 16);
  f.q[0] = *reinterpret_cast<const uint4*>(p);   // ds_load_b128
  f.q[1] = *reinterpret_cast<const uint4*>(p + 8);
  return f.v;
}

__device__ __forceinline__ v8f wmma_bf16(v16bf a, v16bf b, v8f c) {
  return __builtin_amdgcn_wmma_f32_16x16x32_bf16(false, a, false, b, (short)0, c, false, false);
}

__device__ __forceinline__ float sigmoidf_(float x) { return 1.0f / (1.0f + __expf(-x)); }

// ---------------- grid-wide barrier (32 co-resident blocks) ----------------
__device__ __forceinline__ void grid_barrier(unsigned* cnt, unsigned target) {
  __threadfence();
  __syncthreads();
  if (threadIdx.x == 0) {
    __hip_atomic_fetch_add(cnt, 1u, __ATOMIC_ACQ_REL, __HIP_MEMORY_SCOPE_AGENT);
    while (__hip_atomic_load(cnt, __ATOMIC_ACQUIRE, __HIP_MEMORY_SCOPE_AGENT) < target) {
      __builtin_amdgcn_s_sleep(2);
    }
  }
  __syncthreads();
  __threadfence();
}

// ---------------- TDM descriptor helper (CDNA5 ISA, D# groups) ----------------
__device__ __forceinline__ void set_field(uint32_t* d, int lo, int width, unsigned long long v) {
#pragma unroll
  for (int i = 0; i < 8; ++i) {
    int bitlo = i * 32;
    if (lo + width <= bitlo || lo >= bitlo + 32) continue;
    int sh = lo - bitlo;
    unsigned long long chunk = (sh >= 0) ? (v << sh) : (v >> (-sh));
    d[i] |= (uint32_t)chunk;
  }
}

// ---------------- prep kernels ----------------
__global__ void prep_weights(const float* __restrict__ Wih, const float* __restrict__ Whh,
                             const float* __restrict__ bih, const float* __restrict__ bhh,
                             const float* __restrict__ Wout,
                             bf16_t* __restrict__ wg, bf16_t* __restrict__ wo,
                             float* __restrict__ biasg, bf16_t* __restrict__ hb,
                             unsigned* __restrict__ cnt) {
  const int tid = blockIdx.x * blockDim.x + threadIdx.x;
  const int stride = gridDim.x * blockDim.x;
  for (int idx = tid; idx < 2048 * 640; idx += stride) {  // wg[g][c] = [W_ih | W_hh]
    int g = idx / 640, c = idx % 640;
    float v = (c < 128) ? Wih[g * 128 + c] : Whh[g * 512 + (c - 128)];
    wg[idx] = (bf16_t)v;
  }
  for (int idx = tid; idx < 128 * 512; idx += stride) wo[idx] = (bf16_t)Wout[idx];
  for (int idx = tid; idx < 2048; idx += stride) biasg[idx] = bih[idx] + bhh[idx];
  for (int idx = tid; idx < 2 * 64 * 512; idx += stride) hb[idx] = (bf16_t)0.0f;  // h0 = 0
  if (tid == 0) *cnt = 0u;
}

__global__ void prep_inputs(const float* __restrict__ x, bf16_t* __restrict__ xT) {
  const int tid = blockIdx.x * blockDim.x + threadIdx.x;
  const int stride = gridDim.x * blockDim.x;
  for (int idx = tid; idx < 512 * 64 * 128; idx += stride) {  // xT[s][b][k] = x[b][s][k]
    int s = idx / (64 * 128);
    int r = idx % (64 * 128);
    int b = r / 128, k = r % 128;
    xT[idx] = (bf16_t)x[((size_t)b * 512 + s) * 128 + k];
  }
}

// ---------------- persistent fused LSTM scan ----------------
__global__ __launch_bounds__(128, 1)
void lstm_persistent(const bf16_t* __restrict__ xT, const bf16_t* __restrict__ wg,
                     const bf16_t* __restrict__ wo, const float* __restrict__ biasg,
                     const float* __restrict__ bout, bf16_t* __restrict__ hbuf,
                     unsigned* __restrict__ cnt, float* __restrict__ out) {
  extern __shared__ char smem_raw[];                 // 80 KB weight slab
  bf16_t* slab = reinterpret_cast<bf16_t*>(smem_raw);

  const int lane = threadIdx.x & 31;
  const int lrow = lane & 15, lhi = lane >> 4;
  const int mt = threadIdx.x >> 5;                   // wave in block = batch tile (0..3)
  const int nt = blockIdx.x;                         // block = hidden tile (0..31)
  const int mbase = mt * 16, nbase = nt * 16;
  const int n = nbase + lrow;                        // this lane's hidden unit

  // ---- prologue: TDM-stage this block's 64x640 bf16 weight slab into LDS ----
  // rows: 4 gate groups of 16 (nbase+g*512 .. +15), 640 cols each; LDS = [gate][row][640]
#if __has_builtin(__builtin_amdgcn_tensor_load_to_lds)
  if (threadIdx.x < 32) {                            // one wave issues the DMA (EXEC ignored)
    uint64_t gaddr = (uint64_t)(uintptr_t)(wg + (size_t)nbase * 640);
    uint32_t g0[4] = {0, 0, 0, 0};
    g0[0] = 1u;                                      // count=1, user mode, no gather
    g0[1] = (uint32_t)(uintptr_t)smem_raw;           // lds_addr
    g0[2] = (uint32_t)gaddr;                         // global_addr[31:0]
    g0[3] = (uint32_t)((gaddr >> 32) & 0x1FFFFFFu) | (2u << 30);  // addr[56:32] | type=2
    uint32_t g1[8] = {0, 0, 0, 0, 0, 0, 0, 0};
    set_field(g1, 16, 2, 2ull);                      // data_size = 4 bytes
    set_field(g1, 48, 32, 320ull);                   // tensor_dim0  = 320 dwords (one row)
    set_field(g1, 80, 32, 16ull);                    // tensor_dim1  = 16 rows
    set_field(g1, 112, 16, 320ull);                  // tile_dim0    = 320 dwords
    set_field(g1, 128, 16, 16ull);                   // tile_dim1    = 16 rows
    set_field(g1, 144, 16, 4ull);                    // tile_dim2    = 4 gate groups
    set_field(g1, 160, 48, 320ull);                  // tensor_dim0_stride = 320 dwords
    set_field(g1, 208, 48, 512ull * 320ull);         // tensor_dim1_stride = 512 rows
    uint32_t g2[4] = {4u, 0, 0, 0};                  // tensor_dim2 = 4
    uint32_t g3[4] = {0, 0, 0, 0};
    tdm_v4u G0; tdm_v8i G1; tdm_v4i G2, G3; tdm_v8i G4;
#pragma unroll
    for (int i = 0; i < 4; ++i) { G0[i] = g0[i]; G2[i] = (int)g2[i]; G3[i] = (int)g3[i]; }
#pragma unroll
    for (int i = 0; i < 8; ++i) { G1[i] = (int)g1[i]; G4[i] = 0; }
    __builtin_amdgcn_tensor_load_to_lds(G0, G1, G2, G3, G4, 0);
    __builtin_amdgcn_s_wait_tensorcnt(0);
  }
#else
  for (int idx = threadIdx.x; idx < 64 * 640 / 8; idx += blockDim.x) {
    int g = idx / (16 * 80), rem = idx % (16 * 80);
    int r = rem / 80, c = rem % 80;
    reinterpret_cast<uint4*>(slab)[idx] =
        *reinterpret_cast<const uint4*>(wg + (size_t)(nbase + g * 512 + r) * 640 + c * 8);
  }
#endif
  __syncthreads();                                   // slab visible to all 4 waves

  // biases are step-invariant: hoist to registers
  const float bI = biasg[n], bF = biasg[n + 512], bG = biasg[n + 1024], bO = biasg[n + 1536];

  // output-projection duty: 32 tiles (4 M x 8 N over F=128) for blocks 0..7
  const bool outduty = (nt < 8);
  const int omb = mbase;                             // batch tile
  const int onb = nt * 16;                           // feature tile
  const float ob = outduty ? bout[onb + lrow] : 0.0f;

  bf16_t* hb0 = hbuf;
  bf16_t* hb1 = hbuf + 64 * 512;

  v8f c_state = {};  // cell state lives in VGPRs for the whole scan
  int p = 0;

  for (int s = 0; s < 512; ++s) {
    const bf16_t* xs = xT + (size_t)s * (64 * 128);
    const bf16_t* hprev = p ? hb1 : hb0;
    bf16_t* hnext = p ? hb0 : hb1;

    v8f aI = {}, aF = {}, aG = {}, aO = {};

    // K = 0..127 : input part (A from xT[s], stride 128; B from LDS slab)
#pragma unroll
    for (int kt = 0; kt < 4; ++kt) {
      v16bf a = load_frag_a(xs, 128, mbase, kt * 32, lrow, lhi);
      aI = wmma_bf16(a, load_frag_b_lds(slab, 0, kt * 32, lrow, lhi), aI);
      aF = wmma_bf16(a, load_frag_b_lds(slab, 1, kt * 32, lrow, lhi), aF);
      aG = wmma_bf16(a, load_frag_b_lds(slab, 2, kt * 32, lrow, lhi), aG);
      aO = wmma_bf16(a, load_frag_b_lds(slab, 3, kt * 32, lrow, lhi), aO);
    }
    // K = 128..639 : recurrent part (A from h_{s-1}, stride 512; B from LDS slab)
#pragma unroll 4
    for (int kt = 4; kt < 20; ++kt) {
      v16bf a = load_frag_a(hprev, 512, mbase, kt * 32 - 128, lrow, lhi);
      aI = wmma_bf16(a, load_frag_b_lds(slab, 0, kt * 32, lrow, lhi), aI);
      aF = wmma_bf16(a, load_frag_b_lds(slab, 1, kt * 32, lrow, lhi), aF);
      aG = wmma_bf16(a, load_frag_b_lds(slab, 2, kt * 32, lrow, lhi), aG);
      aO = wmma_bf16(a, load_frag_b_lds(slab, 3, kt * 32, lrow, lhi), aO);
    }

    // elementwise gate math + c/h update; write h_s tile (bf16, row-major [64][512])
#pragma unroll
    for (int r = 0; r < 8; ++r) {
      float ig = sigmoidf_(aI[r] + bI);
      float fg = sigmoidf_(aF[r] + bF);
      float gg = tanhf(aG[r] + bG);
      float og = sigmoidf_(aO[r] + bO);
      float cc = fg * c_state[r] + ig * gg;
      c_state[r] = cc;
      float hh = og * tanhf(cc);
      hnext[(size_t)(mbase + lhi * 8 + r) * 512 + n] = (bf16_t)hh;
    }

    if (s + 1 < 512)  // warm L2/WGP$ for next step's input slab
      __builtin_prefetch(xT + (size_t)(s + 1) * (64 * 128) + (size_t)(mbase + lrow) * 128, 0, 0);

    grid_barrier(cnt, (unsigned)(GRID_BLKS * (s + 1)));  // h_s now globally visible

    // out_s = h_s @ W_out^T + b_out  (overlaps with other blocks starting step s+1)
    if (outduty) {
      v8f acc = {};
#pragma unroll 4
      for (int kt = 0; kt < 16; ++kt) {
        v16bf a = load_frag_a(hnext, 512, omb, kt * 32, lrow, lhi);
        acc = wmma_bf16(a, load_frag_b(wo, 512, onb, kt * 32, lrow, lhi), acc);
      }
#pragma unroll
      for (int r = 0; r < 8; ++r) {
        out[(size_t)(omb + lhi * 8 + r) * (512 * 128) + (size_t)s * 128 + onb + lrow] =
            acc[r] + ob;
      }
    }
    p ^= 1;
  }
}

// ---------------- host entry ----------------
extern "C" void kernel_launch(void* const* d_in, const int* in_sizes, int n_in,
                              void* d_out, int out_size, void* d_ws, size_t ws_size,
                              hipStream_t stream) {
  const float* x    = (const float*)d_in[0];  // [64,512,128]
  const float* Wih  = (const float*)d_in[1];  // [2048,128]
  const float* Whh  = (const float*)d_in[2];  // [2048,512]
  const float* bih  = (const float*)d_in[3];  // [2048]
  const float* bhh  = (const float*)d_in[4];  // [2048]
  const float* Wout = (const float*)d_in[5];  // [128,512]
  const float* bout = (const float*)d_in[6];  // [128]

  char* ws = (char*)d_ws;
  unsigned* cnt  = (unsigned*)(ws + CNT_OFF);
  float*   biasg = (float*)(ws + BIAS_OFF);
  bf16_t*  wg    = (bf16_t*)(ws + WG_OFF);
  bf16_t*  wo    = (bf16_t*)(ws + WOUT_OFF);
  bf16_t*  xT    = (bf16_t*)(ws + XT_OFF);
  bf16_t*  hb    = (bf16_t*)(ws + HB_OFF);

  prep_weights<<<64, 256, 0, stream>>>(Wih, Whh, bih, bhh, Wout, wg, wo, biasg, hb, cnt);
  prep_inputs<<<256, 256, 0, stream>>>(x, xT);
  lstm_persistent<<<GRID_BLKS, 128, SLAB_BYTES, stream>>>(xT, wg, wo, biasg, bout, hb, cnt,
                                                          (float*)d_out);
}
